// MultiHeadModulator_90804198572664
// MI455X (gfx1250) — compile-verified, matrix operand order checked by reading, not run on hardware
//
#include <hip/hip_runtime.h>

#define DIMC     2048
#define HEADS    16
#define HEAD_DIM 128
#define PAST_LEN 8192
#define D2       4096
#define SCALING  4.0f

#if defined(__AMDGCN__) && __has_builtin(__builtin_amdgcn_global_load_async_to_lds_b128)
#define HAVE_ASYNC_LDS 1
#else
#define HAVE_ASYNC_LDS 0
#endif

typedef __attribute__((ext_vector_type(16))) __bf16 v16bf;
typedef __attribute__((ext_vector_type(8)))  float  v8f;
typedef __attribute__((ext_vector_type(4)))  int    v4i;

#if HAVE_ASYNC_LDS
typedef __attribute__((address_space(1))) v4i GV4;   // global 128-bit chunk
typedef __attribute__((address_space(3))) v4i LV4;   // LDS 128-bit chunk

static __device__ __forceinline__ GV4* as_global_v4(const void* p) {
    return (GV4*)(unsigned long long)p;              // flat == global numerically
}
static __device__ __forceinline__ LV4* as_lds_v4(const void* p) {
    // flat LDS address: aperture in [63:32], LDS byte offset in [31:0]
    return (LV4*)(unsigned)(unsigned long long)p;
}
static __device__ __forceinline__ void wait_async0() {
#if __has_builtin(__builtin_amdgcn_s_wait_asynccnt)
    __builtin_amdgcn_s_wait_asynccnt(0);
#else
    asm volatile("s_wait_asynccnt 0" ::: "memory");
#endif
}
#endif

static __device__ __forceinline__ float bf2f(unsigned short h) {
    return __uint_as_float(((unsigned)h) << 16);
}
static __device__ __forceinline__ unsigned short f2bf(float f) {
    unsigned u = __float_as_uint(f);
    u += 0x7FFFu + ((u >> 16) & 1u);   // round-to-nearest-even
    return (unsigned short)(u >> 16);
}
static __device__ __forceinline__ float dq4(float w, float inv, float scale) {
    return fminf(fmaxf(rintf(w * inv), -8.0f), 7.0f) * scale;
}

// ---------------- abs-max reduction (fp32 >= 0 orders like uint) ----------------
__global__ void absmax_kernel(const float* __restrict__ w, size_t n4,
                              unsigned* __restrict__ out) {
    __shared__ float red[256];
    const float4* w4 = (const float4*)w;
    float m = 0.f;
    for (size_t i = (size_t)blockIdx.x * 256 + threadIdx.x; i < n4;
         i += (size_t)gridDim.x * 256) {
        float4 v = w4[i];
        m = fmaxf(m, fmaxf(fmaxf(fabsf(v.x), fabsf(v.y)),
                           fmaxf(fabsf(v.z), fabsf(v.w))));
    }
    red[threadIdx.x] = m;
    __syncthreads();
    for (int s = 128; s > 0; s >>= 1) {
        if (threadIdx.x < s) red[threadIdx.x] = fmaxf(red[threadIdx.x], red[threadIdx.x + s]);
        __syncthreads();
    }
    if (threadIdx.x == 0) atomicMax(out, __float_as_uint(red[0]));
}

// ---------------- quantize fp32 -> 4-bit levels -> bf16 ----------------
__global__ void quant_bf16_kernel(const float* __restrict__ w,
                                  const unsigned* __restrict__ amax,
                                  unsigned short* __restrict__ out, size_t n4) {
    const float scale = __uint_as_float(*amax) * (1.0f / 7.0f);
    const float inv   = 1.0f / (scale + 1e-8f);
    const float4* w4 = (const float4*)w;
    ushort4* o4 = (ushort4*)out;
    for (size_t i = (size_t)blockIdx.x * 256 + threadIdx.x; i < n4;
         i += (size_t)gridDim.x * 256) {
        float4 v = w4[i];
        ushort4 r;
        r.x = f2bf(dq4(v.x, inv, scale));
        r.y = f2bf(dq4(v.y, inv, scale));
        r.z = f2bf(dq4(v.z, inv, scale));
        r.w = f2bf(dq4(v.w, inv, scale));
        o4[i] = r;
    }
}

__global__ void quant_f32_kernel(const float* __restrict__ w,
                                 const unsigned* __restrict__ amax,
                                 float* __restrict__ out, size_t n4) {
    const float scale = __uint_as_float(*amax) * (1.0f / 7.0f);
    const float inv   = 1.0f / (scale + 1e-8f);
    const float4* w4 = (const float4*)w;
    float4* o4 = (float4*)out;
    for (size_t i = (size_t)blockIdx.x * 256 + threadIdx.x; i < n4;
         i += (size_t)gridDim.x * 256) {
        float4 v = w4[i];
        float4 r;
        r.x = dq4(v.x, inv, scale); r.y = dq4(v.y, inv, scale);
        r.z = dq4(v.z, inv, scale); r.w = dq4(v.w, inv, scale);
        o4[i] = r;
    }
}

__global__ void f32_to_bf16_kernel(const float* __restrict__ in,
                                   unsigned short* __restrict__ out, size_t n4) {
    const float4* i4 = (const float4*)in;
    ushort4* o4 = (ushort4*)out;
    for (size_t i = (size_t)blockIdx.x * 256 + threadIdx.x; i < n4;
         i += (size_t)gridDim.x * 256) {
        float4 v = i4[i];
        ushort4 r;
        r.x = f2bf(v.x); r.y = f2bf(v.y); r.z = f2bf(v.z); r.w = f2bf(v.w);
        o4[i] = r;
    }
}

// ---------------- WMMA bf16 GEMM: C[p,n] = sum_k A[p,k]*W[n,k] + bias + LoRA ----
// Block tile 128x128, 8 waves (2x4), wave tile 64x32 = 4x2 WMMA accumulators.
// K-tile 64 (2 WMMA K-steps per LDS stage), double-buffered LDS (64 KB).
__launch_bounds__(256)
__global__ void gemm_wmma_kernel(const unsigned short* __restrict__ A,  // [P  x 4096] bf16
                                 const unsigned short* __restrict__ W,  // [4096 x 4096] bf16 (n,k)
                                 const float* __restrict__ qb,          // [4096] quantized bias
                                 const float* __restrict__ t,           // [P x 4] LoRA x@la.T
                                 const float* __restrict__ lb,          // [4096 x 4]
                                 float* __restrict__ C) {               // [P x 4096]
    const int K = D2, N = D2;
    __shared__ __align__(16) unsigned short As[2][128][64];
    __shared__ __align__(16) unsigned short Bs[2][128][64];

    const int tid   = threadIdx.x;
    const int lane  = tid & 31;
    const int wave  = tid >> 5;
    const int waveM = wave >> 2;   // 0..1
    const int waveN = wave & 3;    // 0..3
    const int mBase = blockIdx.y * 128;
    const int nBase = blockIdx.x * 128;

    // tile = 1024 uint4 (128 rows x 64 bf16); 4 uint4 per thread per matrix
    int rr[4], cc[4];
#pragma unroll
    for (int i = 0; i < 4; ++i) {
        const int u = tid + 256 * i;
        rr[i] = u >> 3;
        cc[i] = (u & 7) * 8;
    }

    v8f acc[4][2];
#pragma unroll
    for (int mi = 0; mi < 4; ++mi)
#pragma unroll
        for (int ni = 0; ni < 2; ++ni)
#pragma unroll
            for (int e = 0; e < 8; ++e) acc[mi][ni][e] = 0.0f;

    union Frag { uint4 u[2]; v16bf v; };

    const int KT = K / 64;

#if HAVE_ASYNC_LDS
    auto issue_tile = [&](int kt, int b) {
        const int k0 = kt * 64;
#pragma unroll
        for (int i = 0; i < 4; ++i) {
            __builtin_amdgcn_global_load_async_to_lds_b128(
                as_global_v4(A + (size_t)(mBase + rr[i]) * K + k0 + cc[i]),
                as_lds_v4(&As[b][rr[i]][cc[i]]),
                0, 0);
            __builtin_amdgcn_global_load_async_to_lds_b128(
                as_global_v4(W + (size_t)(nBase + rr[i]) * K + k0 + cc[i]),
                as_lds_v4(&Bs[b][rr[i]][cc[i]]),
                0, 0);
        }
    };
    issue_tile(0, 0);
    wait_async0();
    __syncthreads();
#else
    uint4 pa[4], pb[4];
    auto fetch = [&](int kt) {
        const int k0 = kt * 64;
#pragma unroll
        for (int i = 0; i < 4; ++i) {
            pa[i] = *(const uint4*)(A + (size_t)(mBase + rr[i]) * K + k0 + cc[i]);
            pb[i] = *(const uint4*)(W + (size_t)(nBase + rr[i]) * K + k0 + cc[i]);
        }
    };
    auto stash = [&](int b) {
#pragma unroll
        for (int i = 0; i < 4; ++i) {
            *(uint4*)(&As[b][rr[i]][cc[i]]) = pa[i];
            *(uint4*)(&Bs[b][rr[i]][cc[i]]) = pb[i];
        }
    };
    fetch(0);
    stash(0);
    __syncthreads();
#endif

    const int kb = (lane >> 4) << 3;  // K-half select: lanes 0-15 -> {0-7,16-23}, 16-31 -> {8-15,24-31}
    const int rm = lane & 15;         // row (A) / column (B) within 16

    for (int kt = 0; kt < KT; ++kt) {
        const int cur = kt & 1;
#if HAVE_ASYNC_LDS
        if (kt + 1 < KT) issue_tile(kt + 1, cur ^ 1);
#else
        if (kt + 1 < KT) fetch(kt + 1);
#endif

#pragma unroll
        for (int kk = 0; kk < 64; kk += 32) {
            Frag fa[4], fb[2];
#pragma unroll
            for (int mi = 0; mi < 4; ++mi) {
                const unsigned short* p = &As[cur][waveM * 64 + mi * 16 + rm][kk + kb];
                fa[mi].u[0] = *(const uint4*)(p);
                fa[mi].u[1] = *(const uint4*)(p + 16);
            }
#pragma unroll
            for (int ni = 0; ni < 2; ++ni) {
                const unsigned short* p = &Bs[cur][waveN * 32 + ni * 16 + rm][kk + kb];
                fb[ni].u[0] = *(const uint4*)(p);
                fb[ni].u[1] = *(const uint4*)(p + 16);
            }
#pragma unroll
            for (int mi = 0; mi < 4; ++mi)
#pragma unroll
                for (int ni = 0; ni < 2; ++ni)
                    acc[mi][ni] = __builtin_amdgcn_wmma_f32_16x16x32_bf16(
                        false, fa[mi].v, false, fb[ni].v,
                        (short)0, acc[mi][ni], false, false);
        }

#if HAVE_ASYNC_LDS
        wait_async0();
#else
        if (kt + 1 < KT) stash(cur ^ 1);
#endif
        __syncthreads();
    }

    // Epilogue: C lane layout — VGPR v: lanes 0-15 (M=v, N=lane), 16-31 (M=v+8, N=lane-16)
    const int rowSel = (lane >> 4) << 3;
#pragma unroll
    for (int mi = 0; mi < 4; ++mi) {
        const int m16 = mBase + waveM * 64 + mi * 16 + rowSel;
#pragma unroll
        for (int ni = 0; ni < 2; ++ni) {
            const int n  = nBase + waveN * 32 + ni * 16 + rm;
            const float qbn = qb[n];
            const float l0 = lb[n * 4 + 0], l1 = lb[n * 4 + 1];
            const float l2 = lb[n * 4 + 2], l3 = lb[n * 4 + 3];
#pragma unroll
            for (int v = 0; v < 8; ++v) {
                const int m = m16 + v;
                const float* tp = t + (size_t)m * 4;
                C[(size_t)m * N + n] = acc[mi][ni][v] + qbn +
                    SCALING * (tp[0] * l0 + tp[1] * l1 + tp[2] * l2 + tp[3] * l3);
            }
        }
    }
}

// ---------------- LoRA t[p,r] = kv[p,:] . la[r,:]  (one wave per row) ----------
__global__ void lora_t_kernel(const unsigned short* __restrict__ kvbf,
                              const float* __restrict__ la,  // [4 x 4096]
                              float* __restrict__ t, int P) {
    const int lane = threadIdx.x & 31;
    const int wave = threadIdx.x >> 5;
    const int p = blockIdx.x * 8 + wave;
    if (p >= P) return;
    const unsigned short* row = kvbf + (size_t)p * D2;
    float a0 = 0, a1 = 0, a2 = 0, a3 = 0;
    for (int i = lane * 8; i < D2; i += 256) {
        uint4 u = *(const uint4*)(row + i);
        float x0 = bf2f((unsigned short)(u.x & 0xffff)), x1 = bf2f((unsigned short)(u.x >> 16));
        float x2 = bf2f((unsigned short)(u.y & 0xffff)), x3 = bf2f((unsigned short)(u.y >> 16));
        float x4 = bf2f((unsigned short)(u.z & 0xffff)), x5 = bf2f((unsigned short)(u.z >> 16));
        float x6 = bf2f((unsigned short)(u.w & 0xffff)), x7 = bf2f((unsigned short)(u.w >> 16));
        float4 qa, qc;
        qa = *(const float4*)(la + 0 * D2 + i); qc = *(const float4*)(la + 0 * D2 + i + 4);
        a0 += x0*qa.x + x1*qa.y + x2*qa.z + x3*qa.w + x4*qc.x + x5*qc.y + x6*qc.z + x7*qc.w;
        qa = *(const float4*)(la + 1 * D2 + i); qc = *(const float4*)(la + 1 * D2 + i + 4);
        a1 += x0*qa.x + x1*qa.y + x2*qa.z + x3*qa.w + x4*qc.x + x5*qc.y + x6*qc.z + x7*qc.w;
        qa = *(const float4*)(la + 2 * D2 + i); qc = *(const float4*)(la + 2 * D2 + i + 4);
        a2 += x0*qa.x + x1*qa.y + x2*qa.z + x3*qa.w + x4*qc.x + x5*qc.y + x6*qc.z + x7*qc.w;
        qa = *(const float4*)(la + 3 * D2 + i); qc = *(const float4*)(la + 3 * D2 + i + 4);
        a3 += x0*qa.x + x1*qa.y + x2*qa.z + x3*qa.w + x4*qc.x + x5*qc.y + x6*qc.z + x7*qc.w;
    }
    for (int off = 16; off; off >>= 1) {
        a0 += __shfl_down(a0, off); a1 += __shfl_down(a1, off);
        a2 += __shfl_down(a2, off); a3 += __shfl_down(a3, off);
    }
    if (lane == 0) {
        float* tp = t + (size_t)p * 4;
        tp[0] = a0; tp[1] = a1; tp[2] = a2; tp[3] = a3;
    }
}

// ---------------- rank-4 dots for a single vector ----------------
__global__ void rank_dot_kernel(const float* __restrict__ x,
                                const float* __restrict__ la,
                                float* __restrict__ rt) {
    __shared__ float red[256];
    const int r = blockIdx.x;
    const float* lr = la + (size_t)r * D2;
    float s = 0.f;
    for (int i = threadIdx.x * 4; i < D2; i += 1024) {
        float4 xv = *(const float4*)(x + i);
        float4 lv = *(const float4*)(lr + i);
        s += xv.x * lv.x + xv.y * lv.y + xv.z * lv.z + xv.w * lv.w;
    }
    red[threadIdx.x] = s;
    __syncthreads();
    for (int st = 128; st > 0; st >>= 1) {
        if (threadIdx.x < st) red[threadIdx.x] += red[threadIdx.x + st];
        __syncthreads();
    }
    if (threadIdx.x == 0) rt[r] = red[0];
}

// ---------------- LoRA GEMV with on-the-fly dequant (one wave per row) --------
__global__ void lora_gemv_kernel(const float* __restrict__ x,
                                 const float* __restrict__ w,        // [4096 x 4096]
                                 const unsigned* __restrict__ amax,
                                 const float* __restrict__ qb,
                                 const float* __restrict__ rt,       // [4] = x @ la.T
                                 const float* __restrict__ lbm,      // [4096 x 4]
                                 float* __restrict__ out) {
    const int lane = threadIdx.x & 31;
    const int wave = threadIdx.x >> 5;
    const int j = blockIdx.x * 8 + wave;
    const float scale = __uint_as_float(*amax) * (1.0f / 7.0f);
    const float inv   = 1.0f / (scale + 1e-8f);
    const float* wr = w + (size_t)j * D2;
    float s = 0.f;
    for (int i = lane * 4; i < D2; i += 128) {
        float4 wv = *(const float4*)(wr + i);
        float4 xv = *(const float4*)(x + i);
        s += xv.x * dq4(wv.x, inv, scale) + xv.y * dq4(wv.y, inv, scale) +
             xv.z * dq4(wv.z, inv, scale) + xv.w * dq4(wv.w, inv, scale);
    }
    for (int off = 16; off; off >>= 1) s += __shfl_down(s, off);
    if (lane == 0) {
        const float* lbr = lbm + (size_t)j * 4;
        out[j] = s + qb[j] + SCALING * (rt[0] * lbr[0] + rt[1] * lbr[1] +
                                        rt[2] * lbr[2] + rt[3] * lbr[3]);
    }
}

// ---------------- attention scores: q . (k_p[p] + rel_bias[idx(p)]) ----------
__global__ void scores_kernel(const float* __restrict__ qv,
                              const float* __restrict__ Kp,
                              const float* __restrict__ rel,
                              const int* __restrict__ curr_pos_p,
                              float* __restrict__ scores, int P) {
    const int p = blockIdx.x;
    const int tid = threadIdx.x;
    int idx = p + (curr_pos_p[0] - P) + 64;
    idx = min(max(idx, 0), 128);
    const float* krow = Kp + (size_t)p * D2;
    const float* rrow = rel + (size_t)idx * D2;
    float s = 0.f;
    const int i0 = tid * 16;  // 16 elements per thread, all within one head (256/head)
#pragma unroll
    for (int j = 0; j < 16; j += 4) {
        float4 q4 = *(const float4*)(qv + i0 + j);
        float4 k4 = *(const float4*)(krow + i0 + j);
        float4 r4 = *(const float4*)(rrow + i0 + j);
        s += q4.x * (k4.x + r4.x) + q4.y * (k4.y + r4.y) +
             q4.z * (k4.z + r4.z) + q4.w * (k4.w + r4.w);
    }
    for (int off = 8; off; off >>= 1) s += __shfl_down(s, off, 16);
    if ((tid & 15) == 0) {
        const int h = tid >> 4;
        scores[(size_t)h * P + p] = s * 0.08838834764831845f;  // 1/sqrt(128)
    }
}

// ---------------- per-head softmax over P ----------------
__global__ void softmax_kernel(const float* __restrict__ scores,
                               float* __restrict__ weights, int P) {
    __shared__ float red[256];
    __shared__ float bval;
    const int h = blockIdx.x, tid = threadIdx.x;
    const float* row = scores + (size_t)h * P;
    float m = -3.4e38f;
    for (int p = tid; p < P; p += 256) m = fmaxf(m, row[p]);
    red[tid] = m;
    __syncthreads();
    for (int s = 128; s > 0; s >>= 1) {
        if (tid < s) red[tid] = fmaxf(red[tid], red[tid + s]);
        __syncthreads();
    }
    if (tid == 0) bval = red[0];
    __syncthreads();
    m = bval;
    float z = 0.f;
    for (int p = tid; p < P; p += 256) z += __expf(row[p] - m);
    __syncthreads();
    red[tid] = z;
    __syncthreads();
    for (int s = 128; s > 0; s >>= 1) {
        if (tid < s) red[tid] += red[tid + s];
        __syncthreads();
    }
    if (tid == 0) bval = red[0];
    __syncthreads();
    const float invz = 1.0f / bval;
    float* wrow = weights + (size_t)h * P;
    for (int p = tid; p < P; p += 256) wrow[p] = __expf(row[p] - m) * invz;
}

// ---------------- out_ri[c] += sum_p weights[h,p] * v_p[p,c] (split-P) -------
__global__ void attn_out_kernel(const float* __restrict__ weights,
                                const float* __restrict__ Vp,
                                float* __restrict__ out_ri, int P) {
    const int h = blockIdx.x;
    const int c = h * 256 + threadIdx.x;
    const int p0 = blockIdx.y * 256, p1 = p0 + 256;
    const float* wrow = weights + (size_t)h * P;
    float acc = 0.f;
    for (int p = p0; p < p1; ++p) {
        __builtin_prefetch(Vp + (size_t)(p + 4) * D2 + c, 0, 0);
        acc = fmaf(wrow[p], Vp[(size_t)p * D2 + c], acc);
    }
    atomicAdd(&out_ri[c], acc);
}

// =============================== launcher ====================================
extern "C" void kernel_launch(void* const* d_in, const int* in_sizes, int n_in,
                              void* d_out, int out_size, void* d_ws, size_t ws_size,
                              hipStream_t stream) {
    (void)in_sizes; (void)n_in; (void)out_size; (void)ws_size;
    const float* z_curr   = (const float*)d_in[0];
    const float* z_past   = (const float*)d_in[1];
    const float* rel_bias = (const float*)d_in[2];
    const int*   curr_pos = (const int*)d_in[3];
    const float* w_q = (const float*)d_in[4];  const float* b_q = (const float*)d_in[5];
    const float* la_q = (const float*)d_in[6]; const float* lb_q = (const float*)d_in[7];
    const float* w_k = (const float*)d_in[8];  const float* b_k = (const float*)d_in[9];
    const float* la_k = (const float*)d_in[10]; const float* lb_k = (const float*)d_in[11];
    const float* w_v = (const float*)d_in[12]; const float* b_v = (const float*)d_in[13];
    const float* la_v = (const float*)d_in[14]; const float* lb_v = (const float*)d_in[15];
    const float* w_o = (const float*)d_in[16]; const float* b_o = (const float*)d_in[17];
    const float* la_o = (const float*)d_in[18]; const float* lb_o = (const float*)d_in[19];

    char* base = (char*)d_ws;
    size_t off = 0;
    auto alloc = [&](size_t bytes) -> void* {
        off = (off + 255) & ~(size_t)255;
        void* p = base + off;
        off += bytes;
        return p;
    };

    unsigned* amax = (unsigned*)alloc(8 * sizeof(unsigned)); // wq,wk,wv,wo,bq,bk,bv,bo
    float* qb_q = (float*)alloc(D2 * sizeof(float));
    float* qb_k = (float*)alloc(D2 * sizeof(float));
    float* qb_v = (float*)alloc(D2 * sizeof(float));
    float* qb_o = (float*)alloc(D2 * sizeof(float));
    float* qt     = (float*)alloc(4 * sizeof(float));
    float* ot     = (float*)alloc(4 * sizeof(float));
    float* q_out  = (float*)alloc(D2 * sizeof(float));
    float* out_ri = (float*)alloc(D2 * sizeof(float));
    float* t_k = (float*)alloc((size_t)PAST_LEN * 4 * sizeof(float));
    float* t_v = (float*)alloc((size_t)PAST_LEN * 4 * sizeof(float));
    float* scores  = (float*)alloc((size_t)HEADS * PAST_LEN * sizeof(float));
    float* weights = (float*)alloc((size_t)HEADS * PAST_LEN * sizeof(float));
    unsigned short* kv_bf = (unsigned short*)alloc((size_t)PAST_LEN * D2 * 2);
    unsigned short* wk_bf = (unsigned short*)alloc((size_t)D2 * D2 * 2);
    unsigned short* wv_bf = (unsigned short*)alloc((size_t)D2 * D2 * 2);
    float* k_p = (float*)alloc((size_t)PAST_LEN * D2 * sizeof(float));
    float* v_p = (float*)alloc((size_t)PAST_LEN * D2 * sizeof(float));

    (void)hipMemsetAsync(amax, 0, 8 * sizeof(unsigned), stream);
    (void)hipMemsetAsync(out_ri, 0, D2 * sizeof(float), stream);

    const size_t nW4 = (size_t)D2 * D2 / 4;
    const size_t nB4 = D2 / 4;
    absmax_kernel<<<4096, 256, 0, stream>>>(w_q, nW4, amax + 0);
    absmax_kernel<<<4096, 256, 0, stream>>>(w_k, nW4, amax + 1);
    absmax_kernel<<<4096, 256, 0, stream>>>(w_v, nW4, amax + 2);
    absmax_kernel<<<4096, 256, 0, stream>>>(w_o, nW4, amax + 3);
    absmax_kernel<<<4, 256, 0, stream>>>(b_q, nB4, amax + 4);
    absmax_kernel<<<4, 256, 0, stream>>>(b_k, nB4, amax + 5);
    absmax_kernel<<<4, 256, 0, stream>>>(b_v, nB4, amax + 6);
    absmax_kernel<<<4, 256, 0, stream>>>(b_o, nB4, amax + 7);

    quant_bf16_kernel<<<8192, 256, 0, stream>>>(w_k, amax + 1, wk_bf, nW4);
    quant_bf16_kernel<<<8192, 256, 0, stream>>>(w_v, amax + 2, wv_bf, nW4);
    quant_f32_kernel<<<4, 256, 0, stream>>>(b_q, amax + 4, qb_q, nB4);
    quant_f32_kernel<<<4, 256, 0, stream>>>(b_k, amax + 5, qb_k, nB4);
    quant_f32_kernel<<<4, 256, 0, stream>>>(b_v, amax + 6, qb_v, nB4);
    quant_f32_kernel<<<4, 256, 0, stream>>>(b_o, amax + 7, qb_o, nB4);
    f32_to_bf16_kernel<<<16384, 256, 0, stream>>>(z_past, kv_bf, (size_t)PAST_LEN * D2 / 4);

    rank_dot_kernel<<<4, 256, 0, stream>>>(z_curr, la_q, qt);
    lora_gemv_kernel<<<D2 / 8, 256, 0, stream>>>(z_curr, w_q, amax + 0, qb_q, qt, lb_q, q_out);

    lora_t_kernel<<<PAST_LEN / 8, 256, 0, stream>>>(kv_bf, la_k, t_k, PAST_LEN);
    lora_t_kernel<<<PAST_LEN / 8, 256, 0, stream>>>(kv_bf, la_v, t_v, PAST_LEN);

    dim3 ggrid(D2 / 128, PAST_LEN / 128);
    gemm_wmma_kernel<<<ggrid, 256, 0, stream>>>(kv_bf, wk_bf, qb_k, t_k, lb_k, k_p);
    gemm_wmma_kernel<<<ggrid, 256, 0, stream>>>(kv_bf, wv_bf, qb_v, t_v, lb_v, v_p);

    scores_kernel<<<PAST_LEN, 256, 0, stream>>>(q_out, k_p, rel_bias, curr_pos, scores, PAST_LEN);
    softmax_kernel<<<HEADS, 256, 0, stream>>>(scores, weights, PAST_LEN);
    attn_out_kernel<<<dim3(HEADS, PAST_LEN / 256), 256, 0, stream>>>(weights, v_p, out_ri, PAST_LEN);

    rank_dot_kernel<<<4, 256, 0, stream>>>(out_ri, la_o, ot);
    lora_gemv_kernel<<<D2 / 8, 256, 0, stream>>>(out_ri, w_o, amax + 3, qb_o, ot, lb_o, (float*)d_out);
}
